// Node_GAT_24275155157238
// MI455X (gfx1250) — matrix operand order repaired and checked by reading, not compiled, not run on hardware
//
#include <hip/hip_runtime.h>
#include <hip/hip_bf16.h>

// Node-GAT forward for MI455X (gfx1250, wave32, WMMA).
//
// N=16, M=1024, D=64, H=4. ~19.4 GFLOP -> compute bound; all intermediates
// (K/Q/V^T/dx in f16, 32 MB of d_ws) are L2-resident. Every GEMM runs on
// v_wmma_f32_16x16x32_f16 (f16 in, f32 accumulate).
//
//   1) proj_kernel : K,Q (row-major f16) and V^T (D x M f16) = x@W^T + b
//   2) attn_kernel : per (n,h,16-row i-tile) wave:
//        pass1: S^T tiles = Q_j x K_i^T -> raw row max (lrelu applied once)
//        pass2: recompute S^T, p = exp(lrelu(s)-max); the S^T C-fragment is
//               bit-for-bit the A-fragment of P (in-lane f32->f16 repack);
//               agg += P x V via WMMA, and the softmax row-sum is produced
//               by a 5th WMMA against a constant all-ones B fragment so it
//               lands in accumulator-row layout (no shuffles, no LDS).
//   3) dec_kernel  : out = dx @ Wdec^T + bdec (M x 64 x 256 GEMM), f32 out.

typedef __attribute__((ext_vector_type(16))) _Float16 v16h;
typedef __attribute__((ext_vector_type(8)))  _Float16 v8h;
typedef __attribute__((ext_vector_type(8)))  float    v8f;

static constexpr int NB = 16;    // batch
static constexpr int MM = 1024;  // nodes
static constexpr int DD = 64;    // dim
static constexpr int HH = 4;     // heads
static constexpr float NEG_SLOPE = 0.2f;

__device__ __forceinline__ float lrelu(float s) {
  // slope < 1 => leaky_relu(s) == max(s, slope*s)
  return fmaxf(s, NEG_SLOPE * s);
}

__device__ __forceinline__ v8f zero8() {
  v8f c;
#pragma unroll
  for (int r = 0; r < 8; ++r) c[r] = 0.0f;
  return c;
}

// A-fragment (16x32 f16, per ISA layout): lane L holds row m=L%16,
// halves 0..7 = src[k0..k0+7], halves 8..15 = src[k0+16..k0+23],
// k0 = (L<16 ? 0 : 8). `p` must already point at row + k0.
__device__ __forceinline__ v16h make_a(const _Float16* __restrict__ p) {
  v8h lo = *(const v8h*)(p);
  v8h hi = *(const v8h*)(p + 16);
  v16h a;
#pragma unroll
  for (int e = 0; e < 8; ++e) { a[e] = lo[e]; a[e + 8] = hi[e]; }
  return a;
}

// B-fragment from an f32 row-major B^T source (n rows, k cols):
// lane L holds col n=L%16, halves 0..7 = k0..k0+7, 8..15 = k0+16..k0+23,
// k0 = (L<16 ? 0 : 8). `p` must already point at row + k0.
__device__ __forceinline__ v16h make_b_f32(const float* __restrict__ p) {
  v16h b;
#pragma unroll
  for (int e = 0; e < 8; ++e) { b[e] = (_Float16)p[e]; b[e + 8] = (_Float16)p[e + 16]; }
  return b;
}

__device__ __forceinline__ v8f wmma16(v16h a, v16h b, v8f c) {
  return __builtin_amdgcn_wmma_f32_16x16x32_f16(false, a, false, b,
                                                (short)0, c, false, false);
}

// ---------------------------------------------------------------------------
// Kernel 1: projections. grid 512 x 256. wave = one (n,h, 16-row m-tile).
// ---------------------------------------------------------------------------
__global__ __launch_bounds__(256) void proj_kernel(
    const float* __restrict__ x,
    const float* __restrict__ Wk, const float* __restrict__ bk,
    const float* __restrict__ Wq, const float* __restrict__ bq,
    const float* __restrict__ Wv, const float* __restrict__ bv,
    _Float16* __restrict__ Kf, _Float16* __restrict__ Qf,
    _Float16* __restrict__ Vt) {
  const int lane = threadIdx.x & 31;
  const int gw   = blockIdx.x * 8 + (threadIdx.x >> 5);   // 0..4095
  const int nh   = gw >> 6;                               // 0..63
  const int m0   = (gw & 63) << 4;                        // m-tile base
  const int h    = nh & 3;
  const int n    = nh >> 2;
  const int l15  = lane & 15;
  const int hi16 = (lane & 16) ? 1 : 0;

  // A fragments of x (f32 -> f16), two 32-wide k-chunks over D=64
  v16h ax[2];
  {
    const float* xr = x + ((size_t)(n * MM + m0 + l15)) * DD;
#pragma unroll
    for (int dt = 0; dt < 2; ++dt) {
      const float* p = xr + dt * 32 + (hi16 ? 8 : 0);
      v16h a;
#pragma unroll
      for (int e = 0; e < 8; ++e) {
        a[e]     = (_Float16)p[e];
        a[e + 8] = (_Float16)p[e + 16];
      }
      ax[dt] = a;
    }
  }

  const float* Ws[3] = {Wk + h * DD * DD, Wq + h * DD * DD, Wv + h * DD * DD};
  const float* bs[3] = {bk + h * DD, bq + h * DD, bv + h * DD};

#pragma unroll
  for (int mtx = 0; mtx < 3; ++mtx) {
    const float* W = Ws[mtx];
#pragma unroll
    for (int et = 0; et < 4; ++et) {
      const int e = et * 16 + l15;
      v8f c = zero8();
#pragma unroll
      for (int dt = 0; dt < 2; ++dt) {
        // B^T = W[h] row-major (e, d)
        v16h b = make_b_f32(W + (size_t)e * DD + dt * 32 + hi16 * 8);
        c = wmma16(ax[dt], b, c);
      }
      const float bias = bs[mtx][e];
      if (mtx == 2) {
        // V stored transposed: Vt[(nh*D + e)*M + m]; m = m0 + r + 8*hi16
        // contiguous over r -> one 16B store.
        v8h pk;
#pragma unroll
        for (int r = 0; r < 8; ++r) pk[r] = (_Float16)(c[r] + bias);
        *(v8h*)(Vt + ((size_t)(nh * DD + e)) * MM + m0 + hi16 * 8) = pk;
      } else {
        _Float16* dst = (mtx == 0) ? Kf : Qf;
        // Convert first, then store: lets the stores form a clause instead
        // of interleaving cvt/store with xcnt waits.
        _Float16 vals[8];
#pragma unroll
        for (int r = 0; r < 8; ++r) vals[r] = (_Float16)(c[r] + bias);
#pragma unroll
        for (int r = 0; r < 8; ++r) {
          const int m = m0 + r + hi16 * 8;
          dst[((size_t)(nh * MM + m)) * DD + e] = vals[r];
        }
      }
    }
  }
}

// ---------------------------------------------------------------------------
// Kernel 2: attention. grid 512 x 256. wave = 16 rows i of one (n,h).
// ---------------------------------------------------------------------------
__global__ __launch_bounds__(256) void attn_kernel(
    const float* __restrict__ x,
    const _Float16* __restrict__ Kf, const _Float16* __restrict__ Qf,
    const _Float16* __restrict__ Vt, _Float16* __restrict__ dx2) {
  const int lane = threadIdx.x & 31;
  const int warp = threadIdx.x >> 5;
  const int nh   = blockIdx.x >> 3;                  // 0..63
  const int i0   = (blockIdx.x & 7) * 128 + warp * 16;
  const int n    = nh >> 2;
  const int h    = nh & 3;
  const int l15  = lane & 15;
  const int hi16 = (lane & 16) ? 1 : 0;

  const _Float16* Kbase = Kf + (size_t)nh * MM * DD;
  const _Float16* Qbase = Qf + (size_t)nh * MM * DD;
  const _Float16* Vbase = Vt + (size_t)nh * DD * MM;

  // B fragments of K_i^T (B^T = K rows; n = i, k = d), 2 chunks over D=64
  v16h bK[2];
#pragma unroll
  for (int dt = 0; dt < 2; ++dt)
    bK[dt] = make_a(Kbase + ((size_t)(i0 + l15)) * DD + dt * 32 +
                            hi16 * 8);

  // Constant all-ones B fragment: row-sum generator for softmax.
  v16h onesB;
#pragma unroll
  for (int e = 0; e < 16; ++e) onesB[e] = (_Float16)1.0f;

  // ---- pass 1: raw row max of S (lrelu is monotone -> apply once at end).
  // Lane tracks its own j-subset of row i = i0 + l15; one xor-16 shuffle
  // completes the max.
  float rmax = -3.0e38f;
  for (int jt = 0; jt < MM / 16; ++jt) {
    const _Float16* qr = Qbase + ((size_t)(jt * 16 + l15)) * DD + hi16 * 8;
    v8f c = zero8();
    c = wmma16(make_a(qr), bK[0], c);       // d = 0..31
    c = wmma16(make_a(qr + 32), bK[1], c);  // d = 32..63
#pragma unroll
    for (int r = 0; r < 8; ++r) rmax = fmaxf(rmax, c[r]);
  }
  rmax = fmaxf(rmax, __shfl_xor(rmax, 16, 32));
  rmax = lrelu(rmax);  // = max_j leaky_relu(S[i][j])

  // ---- pass 2: recompute S^T, softmax numerator, P x V accumulate.
  // acc[0..3]: agg d-tiles; acc[4]: row sums (P x ones).
  v8f acc[5];
#pragma unroll
  for (int dt = 0; dt < 5; ++dt) acc[dt] = zero8();

  for (int j0 = 0; j0 < MM; j0 += 32) {
    v16h ap;  // A-fragment of P, filled from two S^T C-fragments in-lane
#pragma unroll
    for (int tt = 0; tt < 2; ++tt) {
      const _Float16* qr =
          Qbase + ((size_t)(j0 + tt * 16 + l15)) * DD + hi16 * 8;
      v8f c = zero8();
      c = wmma16(make_a(qr), bK[0], c);
      c = wmma16(make_a(qr + 32), bK[1], c);
#pragma unroll
      for (int r = 0; r < 8; ++r)
        ap[tt * 8 + r] = (_Float16)__expf(lrelu(c[r]) - rmax);  // p in (0,1]
    }
#pragma unroll
    for (int dt = 0; dt < 4; ++dt) {
      v16h bV = make_a(Vbase + ((size_t)(dt * 16 + l15)) * MM + j0 +
                               hi16 * 8);
      acc[dt] = wmma16(ap, bV, acc[dt]);
    }
    acc[4] = wmma16(ap, onesB, acc[4]);  // row sums, accumulator-row layout
  }

  // Epilogue: normalize, leaky_relu, subtract x, store dx (N, M, H*D) f16.
  // acc rows are m = r + 8*hi16; acc[4][r] is exactly that row's sum.
#pragma unroll
  for (int dt = 0; dt < 4; ++dt) {
    const int d = dt * 16 + l15;
#pragma unroll
    for (int r = 0; r < 8; ++r) {
      const int m = i0 + r + hi16 * 8;
      float v = lrelu(acc[dt][r] / acc[4][r]);
      v -= x[((size_t)(n * MM + m)) * DD + d];
      dx2[((size_t)(n * MM + m)) * (HH * DD) + h * DD + d] = (_Float16)v;
    }
  }
}

// ---------------------------------------------------------------------------
// Kernel 3: decoder. Flat (N*M, 256) x (256, 64) GEMM. grid 128 x 256.
// ---------------------------------------------------------------------------
__global__ __launch_bounds__(256) void dec_kernel(
    const _Float16* __restrict__ dx2, const float* __restrict__ Wdec,
    const float* __restrict__ bdec, float* __restrict__ out) {
  const int lane = threadIdx.x & 31;
  const int gw   = blockIdx.x * 8 + (threadIdx.x >> 5);  // 0..1023
  const int g0   = gw * 16;                              // flat row base
  const int l15  = lane & 15;
  const int hi16 = (lane & 16) ? 1 : 0;

  // Preload the 8 A fragments (k = 0..255 in chunks of 32).
  v16h aF[8];
#pragma unroll
  for (int kt = 0; kt < 8; ++kt)
    aF[kt] = make_a(dx2 + ((size_t)(g0 + l15)) * (HH * DD) + kt * 32 +
                    hi16 * 8);

#pragma unroll
  for (int et = 0; et < 4; ++et) {
    const int e = et * 16 + l15;
    v8f c = zero8();
#pragma unroll
    for (int kt = 0; kt < 8; ++kt) {
      // B^T = Wdec row-major (64, 256)
      v16h b = make_b_f32(Wdec + (size_t)e * (HH * DD) + kt * 32 + hi16 * 8);
      c = wmma16(aF[kt], b, c);
    }
    const float bias = bdec[e];
#pragma unroll
    for (int r = 0; r < 8; ++r)
      out[((size_t)(g0 + r + hi16 * 8)) * DD + e] = c[r] + bias;
  }
}

// ---------------------------------------------------------------------------
extern "C" void kernel_launch(void* const* d_in, const int* in_sizes, int n_in,
                              void* d_out, int out_size, void* d_ws,
                              size_t ws_size, hipStream_t stream) {
  (void)in_sizes; (void)n_in; (void)out_size; (void)ws_size;
  const float* x    = (const float*)d_in[0];
  // d_in[1] = edge, unused by the module
  const float* Wk   = (const float*)d_in[2];
  const float* bk   = (const float*)d_in[3];
  const float* Wq   = (const float*)d_in[4];
  const float* bq   = (const float*)d_in[5];
  const float* Wv   = (const float*)d_in[6];
  const float* bv   = (const float*)d_in[7];
  const float* Wdec = (const float*)d_in[8];
  const float* bdec = (const float*)d_in[9];
  float* out = (float*)d_out;

  // Workspace layout (f16): K, Q, V^T, dx  -> 4 * 8 MB = 32 MB.
  const size_t NHMD = (size_t)NB * HH * MM * DD;  // 4,194,304 elements
  _Float16* Kf  = (_Float16*)d_ws;
  _Float16* Qf  = Kf + NHMD;
  _Float16* Vt  = Qf + NHMD;
  _Float16* dx2 = Vt + NHMD;

  proj_kernel<<<512, 256, 0, stream>>>(x, Wk, bk, Wq, bq, Wv, bv, Kf, Qf, Vt);
  attn_kernel<<<512, 256, 0, stream>>>(x, Kf, Qf, Vt, dx2);
  dec_kernel<<<128, 256, 0, stream>>>(dx2, Wdec, bdec, out);
}
